// VectorQuantizer_28114855919730
// MI455X (gfx1250) — matrix operand order repaired and checked by reading, not compiled, not run on hardware
//
#include <hip/hip_runtime.h>
#include <hip/hip_bf16.h>

typedef __bf16 bf16;
typedef __attribute__((ext_vector_type(16))) __bf16 v16bf;
typedef __attribute__((ext_vector_type(8)))  float  v8f;

// Problem constants
#define NROWS   65536      // 16*64*64 vectors
#define EDIM    64
#define NE      1024
#define ZELEMS  4194304    // 16*64*64*64
#define BSTRIDE 262144     // 64*64*64 (one batch image in floats)
#define CSTRIDE 4096       // 64*64    (one channel plane in floats)

// ---------------------------------------------------------------------------
// Kernel 1: preprocess. 1024 threads: zero counts/sumsq, convert codebook row
// to bf16 (for WMMA B fragments), compute ||e_j||^2 in fp32.
// ---------------------------------------------------------------------------
__global__ void __launch_bounds__(1024)
vq_prep(const float* __restrict__ emb, bf16* __restrict__ emb16,
        float* __restrict__ e2, int* __restrict__ counts,
        float* __restrict__ sumsq) {
  const int t = threadIdx.x;                 // 0..1023 == code index
  counts[t] = 0;
  if (t == 0) *sumsq = 0.0f;
  const float* row = emb + t * EDIM;
  bf16* orow = emb16 + t * EDIM;
  float acc = 0.0f;
#pragma unroll
  for (int k = 0; k < EDIM; ++k) {
    float v = row[k];
    acc += v * v;
    orow[k] = (bf16)v;
  }
  e2[t] = acc;
}

// ---------------------------------------------------------------------------
// Kernel 2: fused distance-GEMM + argmin via v_wmma_f32_16x16x32_bf16.
// Block = 256 threads = 8 waves; each wave owns TWO 16-row M tiles (32 rows)
// -> two independent WMMA accumulator chains per B fragment (halves ds
// traffic per FLOP, fills WMMA->VALU hazard slots with independent work).
// Codebook double-buffered through LDS in 128-code chunks (2 x 16 KB):
// next chunk's global loads are issued before computing the current chunk,
// stores land in the other buffer, one barrier per chunk. Inner tile loop
// capped at unroll 4 to stay under 256 VGPRs (avoids s_set_vgpr_msb churn).
// Running argmin lives in registers so the 65536x1024 score matrix (256 MB)
// never touches memory.
// ---------------------------------------------------------------------------
__global__ void __launch_bounds__(256)
vq_argmin(const float* __restrict__ z, const bf16* __restrict__ emb16,
          const float* __restrict__ e2, int* __restrict__ idx_out) {
  __shared__ __align__(32) bf16 esh[2][128 * EDIM];   // 2 x 16 KB codebook buf
  __shared__ float e2sh[2][128];

  const int tid  = threadIdx.x;
  const int lane = tid & 31;
  const int wave = tid >> 5;
  const int half = lane >> 4;    // which 16-lane half of the wave
  const int mcol = lane & 15;    // M row (for A) / N column (for B,C)

  // ---- Load both A fragments once: 16x32 bf16 A, contiguous K order (same as B).
  // lanes 0-15 hold K {0..15}(+kbase), lanes 16-31 hold K {16..31}.
  const int rowbase = blockIdx.x * 256 + wave * 32;   // 32 rows per wave
  v16bf a0[2], a1[2];
#pragma unroll
  for (int j = 0; j < 2; ++j) {
    const int row = rowbase + j * 16 + mcol;
    const int bb  = row >> 12;           // batch index (4096 hw per batch)
    const int hw  = row & 4095;
    const float* zp = z + (size_t)bb * BSTRIDE + hw; // element k at zp[k*CSTRIDE]
#pragma unroll
    for (int i = 0; i < 8; ++i) {
      a0[j][i]     = (bf16)zp[(size_t)(half * 16 + i) * CSTRIDE];
      a0[j][8 + i] = (bf16)zp[(size_t)(half * 16 + 8 + i) * CSTRIDE];
      a1[j][i]     = (bf16)zp[(size_t)(32 + half * 16 + i) * CSTRIDE];
      a1[j][8 + i] = (bf16)zp[(size_t)(32 + half * 16 + 8 + i) * CSTRIDE];
    }
  }

  float best[2][8];
  int   bidx[2][8];
#pragma unroll
  for (int j = 0; j < 2; ++j)
#pragma unroll
    for (int i = 0; i < 8; ++i) { best[j][i] = 3.402823466e38f; bidx[j][i] = 0; }

  // ---- Stage chunk 0.
  {
    const uint4* g0 = (const uint4*)emb16;
    uint4* l0 = (uint4*)&esh[0][0];
#pragma unroll
    for (int j = 0; j < 4; ++j) l0[tid + j * 256] = g0[tid + j * 256];
    if (tid < 128) e2sh[0][tid] = e2[tid];
    __builtin_prefetch((const void*)(emb16 + (size_t)128 * EDIM
                                     + (size_t)tid * 32), 0, 1);
  }
  __syncthreads();

  for (int chunk = 0; chunk < 8; ++chunk) {
    const bf16*  cur = &esh[chunk & 1][0];
    const float* e2c = &e2sh[chunk & 1][0];

    // Issue next chunk's global loads now; latency hides under the WMMAs.
    uint4 stage[4];
    float e2v = 0.0f;
    if (chunk < 7) {
      const uint4* gsrc = (const uint4*)(emb16 + (size_t)(chunk + 1) * 128 * EDIM);
#pragma unroll
      for (int j = 0; j < 4; ++j) stage[j] = gsrc[tid + j * 256];
      if (tid < 128) e2v = e2[(chunk + 1) * 128 + tid];
      if (chunk < 6)
        __builtin_prefetch((const void*)(emb16 + (size_t)(chunk + 2) * 128 * EDIM
                                         + (size_t)tid * 32), 0, 1);
    }

#pragma unroll 4
    for (int t = 0; t < 8; ++t) {          // 8 code tiles of 16 per chunk
      // B fragment: lane holds column N=mcol, K = half*16 .. +15 (+kbase).
      // emb rows are contiguous over K -> two ds_load_b128 per fragment.
      const bf16* brow = cur + (size_t)(t * 16 + mcol) * EDIM + half * 16;
      v16bf b0 = *(const v16bf*)brow;          // K = 0..31
      v16bf b1 = *(const v16bf*)(brow + 32);   // K = 32..63
      // Two independent accumulator chains (one per M tile).
      v8f c0 = {}, c1 = {};
      c0 = __builtin_amdgcn_wmma_f32_16x16x32_bf16(false, a0[0], false, b0,
                                                   (short)0, c0, false, false);
      c1 = __builtin_amdgcn_wmma_f32_16x16x32_bf16(false, a0[1], false, b0,
                                                   (short)0, c1, false, false);
      c0 = __builtin_amdgcn_wmma_f32_16x16x32_bf16(false, a1[0], false, b1,
                                                   (short)0, c0, false, false);
      c1 = __builtin_amdgcn_wmma_f32_16x16x32_bf16(false, a1[1], false, b1,
                                                   (short)0, c1, false, false);
      const float enorm = e2c[t * 16 + mcol];
      const int   code  = chunk * 128 + t * 16 + mcol;
      // C layout: c[i] = dot(row M=i+8*half, code N=mcol)
#pragma unroll
      for (int i = 0; i < 8; ++i) {
        float s0 = enorm - 2.0f * c0[i];
        float s1 = enorm - 2.0f * c1[i];
        if (s0 < best[0][i]) { best[0][i] = s0; bidx[0][i] = code; }
        if (s1 < best[1][i]) { best[1][i] = s1; bidx[1][i] = code; }
      }
    }

    // Publish next chunk into the other buffer; single barrier per chunk.
    if (chunk < 7) {
      uint4* ldst = (uint4*)&esh[(chunk + 1) & 1][0];
#pragma unroll
      for (int j = 0; j < 4; ++j) ldst[tid + j * 256] = stage[j];
      if (tid < 128) e2sh[(chunk + 1) & 1][tid] = e2v;
      __syncthreads();
    }
  }

  // Cross-lane argmin within each 16-lane half (masks 1..8 never cross halves).
#pragma unroll
  for (int m = 8; m >= 1; m >>= 1) {
#pragma unroll
    for (int j = 0; j < 2; ++j)
#pragma unroll
    for (int i = 0; i < 8; ++i) {
      float os = __shfl_xor(best[j][i], m, 32);
      int   oi = __shfl_xor(bidx[j][i], m, 32);
      if (os < best[j][i] || (os == best[j][i] && oi < bidx[j][i])) {
        best[j][i] = os; bidx[j][i] = oi;
      }
    }
  }
  if (mcol == 0) {   // lane 0 holds rows 0..7, lane 16 holds rows 8..15
#pragma unroll
    for (int j = 0; j < 2; ++j) {
      const int rbase = rowbase + j * 16 + half * 8;
#pragma unroll
      for (int i = 0; i < 8; ++i) idx_out[rbase + i] = bidx[j][i];
    }
  }
}

// ---------------------------------------------------------------------------
// Kernel 3: gather z_q = emb[idx] into (B,C,H,W) output, accumulate
// sum((z_q - z)^2), code histogram, and emit idx as float.
// ---------------------------------------------------------------------------
__global__ void __launch_bounds__(256)
vq_gather(const float* __restrict__ z, const float* __restrict__ emb,
          const int* __restrict__ idx, float* __restrict__ zq,
          float* __restrict__ idx_f, float* __restrict__ sumsq,
          int* __restrict__ counts) {
  const int r  = blockIdx.x * blockDim.x + threadIdx.x;   // 0..65535
  const int id = idx[r];
  idx_f[r] = (float)id;
  atomicAdd(&counts[id], 1);
  const int bb = r >> 12;
  const int hw = r & 4095;
  const float* zp = z   + (size_t)bb * BSTRIDE + hw;
  float*       op = zq  + (size_t)bb * BSTRIDE + hw;
  const float* ep = emb + (size_t)id * EDIM;
  float acc = 0.0f;
#pragma unroll
  for (int c = 0; c < EDIM; ++c) {
    float e = ep[c];
    float d = e - zp[(size_t)c * CSTRIDE];
    acc += d * d;
    op[(size_t)c * CSTRIDE] = e;   // straight-through: forward value == z_q
  }
#pragma unroll
  for (int m = 16; m >= 1; m >>= 1) acc += __shfl_xor(acc, m, 32);
  if ((threadIdx.x & 31) == 0) atomicAdd(sumsq, acc);
}

// ---------------------------------------------------------------------------
// Kernel 4: finalize loss & perplexity (single block).
// ---------------------------------------------------------------------------
__global__ void __launch_bounds__(1024)
vq_final(const int* __restrict__ counts, const float* __restrict__ sumsq,
         float* __restrict__ out2) {
  __shared__ float sh[1024];
  const int t = threadIdx.x;
  const float em = (float)counts[t] * (1.0f / (float)NROWS);
  sh[t] = em * logf(em + 1e-10f);
  __syncthreads();
  for (int s = 512; s > 0; s >>= 1) {
    if (t < s) sh[t] += sh[t + s];
    __syncthreads();
  }
  if (t == 0) {
    out2[0] = 1.25f * (*sumsq) * (1.0f / (float)ZELEMS);  // loss = (1+beta)*MSE
    out2[1] = expf(-sh[0]);                                // perplexity
  }
}

// ---------------------------------------------------------------------------
extern "C" void kernel_launch(void* const* d_in, const int* in_sizes, int n_in,
                              void* d_out, int out_size, void* d_ws, size_t ws_size,
                              hipStream_t stream) {
  const float* z   = (const float*)d_in[0];   // (16,64,64,64) fp32
  const float* emb = (const float*)d_in[1];   // (1024,64) fp32
  float* out = (float*)d_out;
  // d_out layout: [z_q_out: 4194304][loss: 1][perplexity: 1][idx(float): 65536]

  char* ws = (char*)d_ws;
  bf16*  emb16  = (bf16*)(ws + 0);            // 131072 B
  float* e2     = (float*)(ws + 131072);      //   4096 B
  int*   idx    = (int*)(ws + 135168);        // 262144 B
  int*   counts = (int*)(ws + 397312);        //   4096 B
  float* sumsq  = (float*)(ws + 401408);      //      4 B

  vq_prep  <<<1,   1024, 0, stream>>>(emb, emb16, e2, counts, sumsq);
  vq_argmin<<<256,  256, 0, stream>>>(z, emb16, e2, idx);
  vq_gather<<<256,  256, 0, stream>>>(z, emb, idx, out, out + (ZELEMS + 2),
                                      sumsq, counts);
  vq_final <<<1,   1024, 0, stream>>>(counts, sumsq, out + ZELEMS);
}
